// VanillaModel_40690520162688
// MI455X (gfx1250) — compile-verified, matrix operand order checked by reading, not run on hardware
//
#include <hip/hip_runtime.h>

// ---------------------------------------------------------------------------
// Problem sizes (fixed by the reference)
// ---------------------------------------------------------------------------
static constexpr long HH      = 64;
static constexpr long BQ      = 128;      // batch (graphs)
static constexpr long RR      = 131072;   // routers
static constexpr long PP      = 262144;   // packets
static constexpr long E_PASS  = 1048576;
static constexpr long E_TRANS = 1048576;
static constexpr long E_CONN  = 524288;
static constexpr int  NL      = 11;

// ---------------------------------------------------------------------------
// WMMA vector types (CDNA5 / gfx1250, wave32)
// ---------------------------------------------------------------------------
typedef __bf16 v16bf __attribute__((ext_vector_type(16)));
typedef float  v8f   __attribute__((ext_vector_type(8)));

union Frag {
  v16bf    bf;
  unsigned u[8];
};

// two f32 -> packed bf16 pair (lo in [15:0], hi in [31:16]); one v_perm_b32
__device__ __forceinline__ unsigned pack2bf(float lo, float hi) {
  unsigned ul = __float_as_uint(lo) + 0x8000u;   // round-half-up to bf16
  unsigned uh = __float_as_uint(hi) + 0x8000u;
  return __builtin_amdgcn_perm(uh, ul, 0x07060302u);
}

// ---------------------------------------------------------------------------
// Fused GEMM: out[:, coloff:coloff+64] = (res?) + relu(A[nrows,K] @ W[K,64] + b)
// K = KB*32. One wave per 16-row tile. W is converted to bf16 fragment layout
// in LDS once per block; a compiler memory-barrier in the tile loop keeps the
// per-iteration ds reads from being hoisted (register-pressure control).
// ---------------------------------------------------------------------------
template <int KB, bool HAS_RES>
__global__ __launch_bounds__(128) void
k_gemm(const float* __restrict__ A, int lda,
       const float* __restrict__ W,
       const float* __restrict__ bias,
       const float* __restrict__ res,
       float* __restrict__ out, int ldo, int coloff,
       int ntiles) {
  // --- stage W fragments (bf16, WMMA B layout) in LDS ---
  __shared__ unsigned wlds[KB * 4 * 32 * 8];   // KB*4 frags * 32 lanes * 8 dw
  const int nslots = KB * 4 * 32;
  for (int slot = threadIdx.x; slot < nslots; slot += blockDim.x) {
    const int frag  = slot >> 5;
    const int ln    = slot & 31;
    const int kb    = frag >> 2;
    const int nb    = frag & 3;
    const int n     = nb * 16 + (ln & 15);
    const int khalf = (ln >> 4) * 8;
    unsigned* dst = &wlds[slot * 8];
#pragma unroll
    for (int i = 0; i < 8; ++i) {
      const int k = kb * 32 + (i >> 2) * 16 + khalf + (i & 3) * 2;
      dst[i] = pack2bf(W[k * 64 + n], W[(k + 1) * 64 + n]);
    }
  }
  __syncthreads();

  const int lane = threadIdx.x & 31;
  const int wave = threadIdx.x >> 5;
  const int wpb  = blockDim.x >> 5;
  const int m    = lane & 15;     // N for B/C/D, M(row) for A
  const int hi   = lane >> 4;     // K-half (A/B), M-half (C/D)

  // bias per n-block, loaded once (stays in registers across the barrier)
  float bv[4];
#pragma unroll
  for (int nb = 0; nb < 4; ++nb) bv[nb] = bias[nb * 16 + m];

  for (int tile = blockIdx.x * wpb + wave; tile < ntiles;
       tile += gridDim.x * wpb) {
    asm volatile("" ::: "memory");   // keep per-iteration loads in the loop

    // ---- A fragments: 8 consecutive K per half, four b128 loads per block ----
    const int arow = tile * 16 + m;
    Frag af[KB];
#pragma unroll
    for (int kb = 0; kb < KB; ++kb) {
      const int base = kb * 32 + hi * 8;
      const float4* a0 = (const float4*)(A + arow * lda + base);
      const float4* a1 = (const float4*)(A + arow * lda + base + 16);
      const float4 v0 = a0[0], v1 = a0[1], v2 = a1[0], v3 = a1[1];
      af[kb].u[0] = pack2bf(v0.x, v0.y);
      af[kb].u[1] = pack2bf(v0.z, v0.w);
      af[kb].u[2] = pack2bf(v1.x, v1.y);
      af[kb].u[3] = pack2bf(v1.z, v1.w);
      af[kb].u[4] = pack2bf(v2.x, v2.y);
      af[kb].u[5] = pack2bf(v2.z, v2.w);
      af[kb].u[6] = pack2bf(v3.x, v3.y);
      af[kb].u[7] = pack2bf(v3.z, v3.w);
    }

    // ---- 4 output n-blocks; W fragment streamed from LDS per step ----
#pragma unroll
    for (int nb = 0; nb < 4; ++nb) {
      v8f acc = {};
#pragma unroll
      for (int kb = 0; kb < KB; ++kb) {
        Frag wf;
        const uint4* wp = (const uint4*)&wlds[((kb * 4 + nb) * 32 + lane) * 8];
        const uint4 w0 = wp[0], w1 = wp[1];
        wf.u[0] = w0.x; wf.u[1] = w0.y; wf.u[2] = w0.z; wf.u[3] = w0.w;
        wf.u[4] = w1.x; wf.u[5] = w1.y; wf.u[6] = w1.z; wf.u[7] = w1.w;
        acc = __builtin_amdgcn_wmma_f32_16x16x32_bf16(
            false, af[kb].bf, false, wf.bf, (short)0, acc, false, false);
      }
      // ---- fused bias + ReLU (+ residual); C/D layout: N=m, M=r+hi*8 ----
      const int n = nb * 16 + m;
#pragma unroll
      for (int r = 0; r < 8; ++r) {
        const int orow = tile * 16 + r + hi * 8;
        float v = fmaxf(acc[r] + bv[nb], 0.0f);
        if constexpr (HAS_RES) v += res[orow * 64 + n];
        out[orow * ldo + coloff + n] = v;
      }
    }
  }
}

// ---------------------------------------------------------------------------
// Small elementwise / tiny-K kernels
// ---------------------------------------------------------------------------
__global__ void k_fill4(float4* p, int n4) {
  int t = blockIdx.x * blockDim.x + threadIdx.x;
  if (t < n4) p[t] = make_float4(0.f, 0.f, 0.f, 0.f);
}

// r_mid[r,c] = relu(sum_j op[r,j]*Wo[j,c] + bo[c]),  K=4
__global__ void k_router_mid(const float* __restrict__ op,
                             const float* __restrict__ Wo,
                             const float* __restrict__ bo,
                             float* __restrict__ outp, int n) {
  int t = blockIdx.x * blockDim.x + threadIdx.x;
  if (t >= n) return;
  const int r = t >> 6;
  const int c = t & 63;
  float acc = bo[c];
#pragma unroll
  for (int j = 0; j < 4; ++j) acc += op[r * 4 + j] * Wo[j * 64 + c];
  outp[t] = fmaxf(acc, 0.0f);
}

// pcat[p, c] = relu(freq[p]*Wf[c] + bf[c])  (outer product, K=1)
__global__ void k_freq(const float* __restrict__ freq,
                       const float* __restrict__ Wf,
                       const float* __restrict__ bf,
                       float* __restrict__ pcat, int n) {
  int t = blockIdx.x * blockDim.x + threadIdx.x;
  if (t >= n) return;
  const int p = t >> 6;
  const int c = t & 63;
  pcat[p * 128 + c] = fmaxf(freq[p] * Wf[c] + bf[c], 0.0f);
}

// Edge scatter-add: 16 lanes per edge, float4 gather + 4 f32 atomics.
template <bool COUNT>
__global__ void k_agg(const float* __restrict__ feat,
                      const int* __restrict__ esrc,
                      const int* __restrict__ edst,
                      float* __restrict__ outp,
                      int nedges, int ld_src, int ld_out, int coloff,
                      float* __restrict__ cnt) {
  int t = blockIdx.x * blockDim.x + threadIdx.x;
  const int e  = t >> 4;
  const int c4 = (t & 15) << 2;
  if (e >= nedges) return;
  const int s = esrc[e];
  const int d = edst[e];
  const float4 v = *(const float4*)(feat + s * ld_src + c4);
  float* o = outp + d * ld_out + coloff + c4;
  atomicAdd(o + 0, v.x);
  atomicAdd(o + 1, v.y);
  atomicAdd(o + 2, v.z);
  atomicAdd(o + 3, v.w);
  if constexpr (COUNT) {
    if (c4 == 0) atomicAdd(cnt + d, 1.0f);
  }
}

// pmsum[p, :] /= max(cnt[p], 1)
__global__ void k_scale(float* __restrict__ msum,
                        const float* __restrict__ cnt, int n) {
  int t = blockIdx.x * blockDim.x + threadIdx.x;
  if (t >= n) return;
  msum[t] = msum[t] / fmaxf(cnt[t >> 6], 1.0f);
}

// Per-graph readout over contiguous 1024-router blocks: sum + max -> emb[B,128]
__global__ void k_readout(const float* __restrict__ r_feat,
                          float* __restrict__ emb) {
  const int g    = blockIdx.x;
  const int c    = threadIdx.x & 63;
  const int part = threadIdx.x >> 6;   // 0..3
  const float* base = r_feat + g * 1024 * 64;
  float s = 0.0f, mx = -3.4e38f;
  for (int i = 0; i < 256; ++i) {
    const float v = base[(part * 256 + i) * 64 + c];
    s += v;
    mx = fmaxf(mx, v);
  }
  __shared__ float ssum[4][64];
  __shared__ float smax[4][64];
  ssum[part][c] = s;
  smax[part][c] = mx;
  __syncthreads();
  if (part == 0) {
    for (int q = 1; q < 4; ++q) {
      s  += ssum[q][c];
      mx  = fmaxf(mx, smax[q][c]);
    }
    emb[g * 128 + c]      = s;
    emb[g * 128 + 64 + c] = mx;
  }
}

// pred[b, j] = h[b,:] @ h3W[:, j] + h3b[j]   (64 -> 11)
__global__ void k_head3(const float* __restrict__ h,
                        const float* __restrict__ W,
                        const float* __restrict__ b,
                        float* __restrict__ outp, int nb) {
  int t = blockIdx.x * blockDim.x + threadIdx.x;
  if (t >= nb * NL) return;
  const int bi = t / NL, j = t % NL;
  float acc = b[j];
#pragma unroll 8
  for (int k = 0; k < 64; ++k) acc += h[bi * 64 + k] * W[k * NL + j];
  outp[t] = acc;
}

// ---------------------------------------------------------------------------
// Host-side launch helpers
// ---------------------------------------------------------------------------
static inline int nblk(long n, int tb = 256) { return (int)((n + tb - 1) / tb); }

template <int KB, bool HAS_RES>
static void launch_gemm(const float* A, int lda, const float* W,
                        const float* bias, const float* res,
                        float* out, int ldo, int coloff, long nrows,
                        hipStream_t s) {
  const int ntiles = (int)(nrows >> 4);
  int blocks = (ntiles + 3) / 4;                 // 4 waves (128 thr) per block
  if (blocks > 4096) blocks = 4096;
  if (blocks < 1) blocks = 1;
  hipLaunchKernelGGL((k_gemm<KB, HAS_RES>), dim3(blocks), dim3(128), 0, s,
                     A, lda, W, bias, res, out, ldo, coloff, ntiles);
}

// ---------------------------------------------------------------------------
// Entry point
// ---------------------------------------------------------------------------
extern "C" void kernel_launch(void* const* d_in, const int* in_sizes, int n_in,
                              void* d_out, int out_size, void* d_ws,
                              size_t ws_size, hipStream_t stream) {
  (void)in_sizes; (void)n_in; (void)out_size; (void)ws_size;

  const float* freq     = (const float*)d_in[0];
  const float* flit     = (const float*)d_in[1];
  const float* op_type  = (const float*)d_in[2];
  const int* pass_src     = (const int*)d_in[3];
  const int* pass_dst     = (const int*)d_in[4];
  const int* transfer_src = (const int*)d_in[5];
  const int* transfer_dst = (const int*)d_in[6];
  const int* connect_src  = (const int*)d_in[7];
  const int* connect_dst  = (const int*)d_in[8];
  // d_in[9] = router_gid: contiguous repeat(arange(B),1024) -> implicit
  const float* Wf  = (const float*)d_in[10];
  const float* bf  = (const float*)d_in[11];
  const float* Wl  = (const float*)d_in[12];
  const float* bl  = (const float*)d_in[13];
  const float* Wo  = (const float*)d_in[14];
  const float* bo  = (const float*)d_in[15];
  const float* Wfh = (const float*)d_in[16];
  const float* bfh = (const float*)d_in[17];
  const float* Wfn = (const float*)d_in[18];
  const float* bfn = (const float*)d_in[19];
  const float* cWr[2] = {(const float*)d_in[20], (const float*)d_in[24]};
  const float* cbr[2] = {(const float*)d_in[21], (const float*)d_in[25]};
  const float* cWp[2] = {(const float*)d_in[22], (const float*)d_in[26]};
  const float* cbp[2] = {(const float*)d_in[23], (const float*)d_in[27]};
  const float* h1W = (const float*)d_in[28];
  const float* h1b = (const float*)d_in[29];
  const float* h2W = (const float*)d_in[30];
  const float* h2b = (const float*)d_in[31];
  const float* h3W = (const float*)d_in[32];
  const float* h3b = (const float*)d_in[33];
  float* outp = (float*)d_out;

  // ---- workspace carve ----
  float* ws      = (float*)d_ws;
  float* p_feat  = ws;                       // P x 64
  float* r_feat  = p_feat + PP * HH;         // R x 64
  float* scratch = r_feat + RR * HH;         // P x 128 (reused 3 ways)
  float* pcnt    = scratch + PP * 2 * HH;    // P   (directly after scratch)
  float* emb     = pcnt + PP;                // B x 128
  float* hbuf1   = emb + BQ * 2 * HH;        // B x 64
  float* hbuf2   = hbuf1 + BQ * HH;          // B x 64

  // ================= FeatureGen =================
  // routers: r_mid = relu(op @ Wo + bo);  r_feat = relu(r_mid @ Wfn + bfn)
  k_router_mid<<<nblk(RR * HH), 256, 0, stream>>>(op_type, Wo, bo, scratch,
                                                  (int)(RR * HH));
  launch_gemm<2, false>(scratch, 64, Wfn, bfn, nullptr, r_feat, 64, 0, RR,
                        stream);

  // packets: pcat = [relu(freq*Wf+bf) | relu(flit@Wl+bl)]; p_feat = relu(pcat@Wfh+bfh)
  k_freq<<<nblk(PP * HH), 256, 0, stream>>>(freq, Wf, bf, scratch,
                                            (int)(PP * HH));
  launch_gemm<1, false>(flit, 32, Wl, bl, nullptr, scratch, 128, 64, PP,
                        stream);
  launch_gemm<4, false>(scratch, 128, Wfh, bfh, nullptr, p_feat, 64, 0, PP,
                        stream);

  // ================= 2 hetero conv layers =================
  for (int l = 0; l < 2; ++l) {
    float* hcat  = scratch;               // R x 128  ([h1 | h2])
    float* pmsum = scratch + RR * 2 * HH; // P x 64
    // zero hcat + pmsum + pcnt (contiguous: R*128 + P*64 == P*128, then P)
    k_fill4<<<nblk((PP * 2 * HH + PP) / 4), 256, 0, stream>>>(
        (float4*)scratch, (int)((PP * 2 * HH + PP) / 4));
    // pass: packet -> router sum (cols 0..63)
    k_agg<false><<<nblk(E_PASS * 16), 256, 0, stream>>>(
        p_feat, pass_src, pass_dst, hcat, (int)E_PASS, 64, 128, 0, nullptr);
    // connect: router -> router sum (cols 64..127)
    k_agg<false><<<nblk(E_CONN * 16), 256, 0, stream>>>(
        r_feat, connect_src, connect_dst, hcat, (int)E_CONN, 64, 128, 64,
        nullptr);
    // transfer: router -> packet sum + count
    k_agg<true><<<nblk(E_TRANS * 16), 256, 0, stream>>>(
        r_feat, transfer_src, transfer_dst, pmsum, (int)E_TRANS, 64, 64, 0,
        pcnt);
    k_scale<<<nblk(PP * HH), 256, 0, stream>>>(pmsum, pcnt, (int)(PP * HH));
    // r_feat += relu(hcat @ Wr + br)   (in-place residual)
    launch_gemm<4, true>(hcat, 128, cWr[l], cbr[l], r_feat, r_feat, 64, 0, RR,
                         stream);
    // p_feat += relu(pmean @ Wp + bp)  (in-place residual)
    launch_gemm<2, true>(pmsum, 64, cWp[l], cbp[l], p_feat, p_feat, 64, 0, PP,
                         stream);
  }

  // ================= readout + head =================
  k_readout<<<dim3((int)BQ), dim3(256), 0, stream>>>(r_feat, emb);
  launch_gemm<4, false>(emb, 128, h1W, h1b, nullptr, hbuf1, 64, 0, BQ, stream);
  launch_gemm<2, false>(hbuf1, 64, h2W, h2b, nullptr, hbuf2, 64, 0, BQ, stream);
  k_head3<<<nblk(BQ * NL), 256, 0, stream>>>(hbuf2, h3W, h3b, outp, (int)BQ);
}
